// TRANSFORMER_CONV_22668837388780
// MI455X (gfx1250) — matrix operand order repaired and checked by reading, not compiled, not run on hardware
//
#include <hip/hip_runtime.h>
#include <math.h>

// ---------------------------------------------------------------------------
// Types for CDNA5 WMMA (wave32): V_WMMA_F32_16X16X32_BF16
// ---------------------------------------------------------------------------
typedef __attribute__((ext_vector_type(16))) __bf16 bf16x16;
typedef __attribute__((ext_vector_type(8)))  __bf16 bf16x8;
typedef __attribute__((ext_vector_type(8)))  float  f32x8;

static __device__ __forceinline__ __bf16 f2bf(float f) {
  unsigned u = __builtin_bit_cast(unsigned, f);
  unsigned r = (u + 0x7FFFu + ((u >> 16) & 1u)) >> 16;   // round-nearest-even
  return __builtin_bit_cast(__bf16, (unsigned short)r);
}

static inline int cdiv(int a, int b) { return (a + b - 1) / b; }

// ---------------------------------------------------------------------------
// Small utility kernels
// ---------------------------------------------------------------------------
__global__ void zero_i32_kernel(int* __restrict__ p, int n) {
  int i = blockIdx.x * blockDim.x + threadIdx.x;
  if (i < n) p[i] = 0;
}

__global__ void copy_i32_kernel(const int* __restrict__ s, int* __restrict__ d, int n) {
  int i = blockIdx.x * blockDim.x + threadIdx.x;
  if (i < n) d[i] = s[i];
}

__global__ void count_deg_kernel(const int* __restrict__ ei, int E, int* __restrict__ counts) {
  int e = blockIdx.x * blockDim.x + threadIdx.x;
  if (e < E) atomicAdd(&counts[ei[E + e]], 1);   // dst = edge_index[1][e]
}

// Single-block exclusive scan over counts -> rowptr[0..n]
__global__ void scan_block_kernel(const int* __restrict__ counts, int* __restrict__ rowptr, int n) {
  __shared__ int wsums[32];
  __shared__ int carry;
  int lane = threadIdx.x & 31;
  int wave = threadIdx.x >> 5;
  if (threadIdx.x == 0) { carry = 0; rowptr[0] = 0; }
  __syncthreads();
  for (int base = 0; base < n; base += (int)blockDim.x) {
    int i = base + (int)threadIdx.x;
    int v = (i < n) ? counts[i] : 0;
    int x = v;
    for (int d = 1; d < 32; d <<= 1) {
      int t = __shfl_up(x, d, 32);
      if (lane >= d) x += t;
    }
    if (lane == 31) wsums[wave] = x;
    __syncthreads();
    if (wave == 0) {
      int y = wsums[lane];
      for (int d = 1; d < 32; d <<= 1) {
        int t = __shfl_up(y, d, 32);
        if (lane >= d) y += t;
      }
      wsums[lane] = y;
    }
    __syncthreads();
    int pre  = (wave > 0) ? wsums[wave - 1] : 0;
    int incl = x + pre + carry;
    if (i < n) rowptr[i + 1] = incl;
    __syncthreads();
    if (threadIdx.x == blockDim.x - 1) carry = incl;
    __syncthreads();
  }
}

__global__ void csr_fill_kernel(const int* __restrict__ ei, int E,
                                int* __restrict__ cursor, int* __restrict__ csr_src) {
  int e = blockIdx.x * blockDim.x + threadIdx.x;
  if (e >= E) return;
  int src = ei[e];
  int dst = ei[E + e];
  int pos = atomicAdd(&cursor[dst], 1);
  csr_src[pos] = src;
}

// x[N,50] fp32 -> Xb[N,64] bf16 (zero padded K)
__global__ void pack_x_kernel(const float* __restrict__ x, __bf16* __restrict__ xb, int total) {
  int i = blockIdx.x * blockDim.x + threadIdx.x;
  if (i >= total) return;
  int row = i >> 6, c = i & 63;
  xb[i] = f2bf(c < 50 ? x[row * 50 + c] : 0.f);
}

// Pack W{q,k,v,skip} [Fin,160] fp32 -> fragment-native bf16:
// Bp[((kb32*640)+ncol)*32 + klocal], klocal = K within 32-block.
__global__ void pack_w_kernel(const float* __restrict__ wq, const float* __restrict__ wk,
                              const float* __restrict__ wv, const float* __restrict__ ws,
                              __bf16* __restrict__ bp, int fin, int total) {
  int i = blockIdx.x * blockDim.x + threadIdx.x;
  if (i >= total) return;
  int kb32 = i / (640 * 32);
  int rem  = i - kb32 * (640 * 32);
  int ncol = rem >> 5;
  int kl   = rem & 31;
  int krow = kb32 * 32 + kl;
  int which = ncol / 160, col = ncol - which * 160;
  const float* W = which == 0 ? wq : which == 1 ? wk : which == 2 ? wv : ws;
  bp[i] = f2bf(krow < fin ? W[krow * 160 + col] : 0.f);
}

__global__ void pack_bias_kernel(const float* __restrict__ bq, const float* __restrict__ bk,
                                 const float* __restrict__ bv, const float* __restrict__ bs,
                                 float* __restrict__ b640) {
  int i = blockIdx.x * blockDim.x + threadIdx.x;
  if (i >= 640) return;
  int which = i / 160, c = i - which * 160;
  const float* b = which == 0 ? bq : which == 1 ? bk : which == 2 ? bv : bs;
  b640[i] = b[c];
}

// weight-norm column scales: sc[c] = g[c] / ||v[:,c]||
__global__ void wn_scale_kernel(const float* __restrict__ v, const float* __restrict__ g,
                                float* __restrict__ sc, int rows, int cols) {
  int c = blockIdx.x * blockDim.x + threadIdx.x;
  if (c >= cols) return;
  float s = 0.f;
  for (int r = 0; r < rows; ++r) { float a = v[r * cols + c]; s += a * a; }
  sc[c] = g[c] / sqrtf(s);
}

// ---------------------------------------------------------------------------
// WMMA GEMM: C[M,640] = A[M,K]bf16 @ Bp(K,640 packed)bf16 + bias.
// One wave computes a 16x64 strip of C; 4 accumulators share one A fragment.
// ---------------------------------------------------------------------------
__global__ __launch_bounds__(256)
void gemm_wmma_kernel(const __bf16* __restrict__ A, const __bf16* __restrict__ Bp,
                      const float* __restrict__ bias, float* __restrict__ C,
                      int M, int K) {
  const int Nout = 640;
  int wid  = (blockIdx.x * blockDim.x + threadIdx.x) >> 5;
  int lane = threadIdx.x & 31;
  int mt = wid / (Nout >> 6);
  int nt = wid - mt * (Nout >> 6);
  int m0 = mt * 16, n0 = nt * 64;
  if (m0 + 16 > M) return;

  const int aoff  = (lane & 16);           // 0 or 16: K half for A fragment
  const int bhalf = (lane & 16);           // 0 or 16: K half for B fragment
  const __bf16* arow = A + (size_t)(m0 + (lane & 15)) * K;

  f32x8 acc0 = {}, acc1 = {}, acc2 = {}, acc3 = {};
  for (int kb = 0; kb < K; kb += 32) {
    // A fragment: 16 contiguous K values per half-wave (matches B)
    bf16x8 alo = *(const bf16x8*)(arow + kb + aoff);
    bf16x8 ahi = *(const bf16x8*)(arow + kb + 8 + aoff);
    bf16x16 af = __builtin_shufflevector(alo, ahi, 0, 1, 2, 3, 4, 5, 6, 7,
                                         8, 9, 10, 11, 12, 13, 14, 15);
    // B fragments: column = n0 + 16j + (lane&15); 16 contiguous K values per half-wave
    const __bf16* bcol = Bp + (((size_t)(kb >> 5) * Nout + n0 + (lane & 15)) << 5) + bhalf;
    bf16x16 b0 = *(const bf16x16*)(bcol);
    bf16x16 b1 = *(const bf16x16*)(bcol + 16 * 32);
    bf16x16 b2 = *(const bf16x16*)(bcol + 32 * 32);
    bf16x16 b3 = *(const bf16x16*)(bcol + 48 * 32);
    acc0 = __builtin_amdgcn_wmma_f32_16x16x32_bf16(false, af, false, b0, (short)0, acc0, false, false);
    acc1 = __builtin_amdgcn_wmma_f32_16x16x32_bf16(false, af, false, b1, (short)0, acc1, false, false);
    acc2 = __builtin_amdgcn_wmma_f32_16x16x32_bf16(false, af, false, b2, (short)0, acc2, false, false);
    acc3 = __builtin_amdgcn_wmma_f32_16x16x32_bf16(false, af, false, b3, (short)0, acc3, false, false);
  }
  // C/D layout: M = m0 + r + (lane>=16 ? 8 : 0), N = n0 + 16j + (lane&15)
  int crow = (lane & 16) ? 8 : 0;
  int ccol = lane & 15;
  f32x8 accs[4] = {acc0, acc1, acc2, acc3};
  for (int j = 0; j < 4; ++j) {
    int n = n0 + j * 16 + ccol;
    float bv = bias[n];
    float* cp = C + (size_t)(m0 + crow) * Nout + n;
    for (int r = 0; r < 8; ++r) cp[(size_t)r * Nout] = accs[j][r] + bv;
  }
}

// ---------------------------------------------------------------------------
// TransformerConv edge attention: one wave per destination node.
// Lanes 0-15: head 0 (5 channels each), lanes 16-31: head 1. Online softmax.
// Epilogue: relu(agg + skip), write fp32 (for pool) and bf16 (next GEMM).
// ---------------------------------------------------------------------------
__global__ __launch_bounds__(256)
void attn_kernel(const float* __restrict__ qkvs, const int* __restrict__ rowptr,
                 const int* __restrict__ csr_src, float* __restrict__ hf,
                 __bf16* __restrict__ hb, int n) {
  int wid  = (blockIdx.x * blockDim.x + threadIdx.x) >> 5;
  int lane = threadIdx.x & 31;
  if (wid >= n) return;
  int head = lane >> 4;
  int lh   = lane & 15;
  int cbase = head * 80 + lh * 5;

  const float* qrow = qkvs + (size_t)wid * 640;
  float qc[5];
  for (int j = 0; j < 5; ++j) qc[j] = qrow[cbase + j];

  const float scale = 0.1118033988749895f;  // 1/sqrt(80)
  float m = -__builtin_inff(), s = 0.f;
  float a[5] = {0.f, 0.f, 0.f, 0.f, 0.f};

  int beg = rowptr[wid], end = rowptr[wid + 1];
  for (int e = beg; e < end; ++e) {
    int src = csr_src[e];
    const float* kr = qkvs + (size_t)src * 640 + 160;
    float p = 0.f;
    for (int j = 0; j < 5; ++j) p += qc[j] * kr[cbase + j];
    p += __shfl_xor(p, 1, 32);
    p += __shfl_xor(p, 2, 32);
    p += __shfl_xor(p, 4, 32);
    p += __shfl_xor(p, 8, 32);           // reduced within 16-lane half (per head)
    float score = p * scale;
    float nm = fmaxf(m, score);
    float rs = __expf(m - nm);           // 0 on first edge (m = -inf)
    float w  = __expf(score - nm);
    s = s * rs + w;
    const float* vr = qkvs + (size_t)src * 640 + 320;
    for (int j = 0; j < 5; ++j) a[j] = a[j] * rs + w * vr[cbase + j];
    m = nm;
  }
  float inv = (s == 0.f) ? 1.f : 1.f / s;
  const float* sk = qrow + 480;
  size_t obase = (size_t)wid * 160 + cbase;
  for (int j = 0; j < 5; ++j) {
    float o = fmaxf(a[j] * inv + sk[cbase + j], 0.f);  // fused conv relu
    hf[obase + j] = o;
    hb[obase + j] = f2bf(o);
  }
}

// Global max pool using int atomicMax (valid: post-relu values are >= 0)
__global__ void pool_kernel(const float* __restrict__ hf, const int* __restrict__ bidx,
                            int* __restrict__ pool, int total) {
  int i = blockIdx.x * blockDim.x + threadIdx.x;
  if (i >= total) return;
  int node = i / 160, c = i - node * 160;
  atomicMax(&pool[bidx[node] * 160 + c], __float_as_int(hf[i]));
}

// ---------------------------------------------------------------------------
// Per-graph MLP head (one workgroup of 256 per graph)
// ---------------------------------------------------------------------------
struct HeadArgs {
  const float *h1W, *h1b, *h2W, *h2b, *h3W, *h3b, *h4W, *h4b;
  const float *s1v1, *s1b1, *s1v2, *s1b2;
  const float *s2v1, *s2b1, *s2v2, *s2b2;
  const float *s3v1, *s3b1, *s3v2, *s3b2;
  const float *s4v1, *s4b1, *s4v2, *s4b2;
  const float *sc1a, *sc1b, *sc2a, *sc2b, *sc3a, *sc3b, *sc4a, *sc4b;
  const float *fwdW, *fwdb, *bwdW, *bwdb, *gateW, *gateb;
};

__global__ __launch_bounds__(256)
void head_kernel(HeadArgs P, const int* __restrict__ pool, const float* __restrict__ tm,
                 float* __restrict__ out) {
  __shared__ float xa[320];
  __shared__ float xb[320];
  __shared__ float bwds[1000];
  const int t = threadIdx.x;
  const int g = blockIdx.x;

  if (t < 160) xa[t] = __int_as_float(pool[g * 160 + t]);
  __syncthreads();

  auto lin_relu = [&](const float* W, const float* b, int in, int outd) {
    for (int c = t; c < outd; c += 256) {
      float acc = b[c];
      for (int k = 0; k < in; ++k) acc += xa[k] * W[k * outd + c];
      xb[c] = fmaxf(acc, 0.f);
    }
    __syncthreads();
    for (int c = t; c < outd; c += 256) xa[c] = xb[c];
    __syncthreads();
  };
  auto skip_block = [&](const float* v1, const float* b1, const float* sc1,
                        const float* v2, const float* b2, const float* sc2,
                        int in, int mid) {
    for (int c = t; c < mid; c += 256) {
      float d = 0.f;
      for (int k = 0; k < in; ++k) d += xa[k] * v1[k * mid + c];
      xb[c] = fmaxf(d * sc1[c] + b1[c], 0.f);
    }
    __syncthreads();
    for (int c = t; c < in; c += 256) {
      float d = 0.f;
      for (int k = 0; k < mid; ++k) d += xb[k] * v2[k * in + c];
      xa[c] = fmaxf(d * sc2[c] + b2[c] + xa[c], 0.f);
    }
    __syncthreads();
  };

  lin_relu(P.h1W, P.h1b, 160, 160);
  skip_block(P.s1v1, P.s1b1, P.sc1a, P.s1v2, P.s1b2, P.sc1b, 160, 240);
  lin_relu(P.h2W, P.h2b, 160, 240);
  skip_block(P.s2v1, P.s2b1, P.sc2a, P.s2v2, P.s2b2, P.sc2b, 240, 160);
  lin_relu(P.h3W, P.h3b, 240, 320);
  skip_block(P.s3v1, P.s3b1, P.sc3a, P.s3v2, P.s3b2, P.sc3b, 320, 160);
  lin_relu(P.h4W, P.h4b, 320, 160);
  skip_block(P.s4v1, P.s4b1, P.sc4a, P.s4v2, P.s4b2, P.sc4b, 160, 160);

  // Output heads with mass-shift mask / reversal / gating
  const int m = (int)rintf(tm[g]);  // round-nearest-even, as jnp.round
  float fr[4], gr[4];
  for (int it = 0; it < 4; ++it) {
    int c = t + it * 256;
    float fa = 0.f, ba = 0.f, ga = 0.f;
    if (c < 1000) {
      fa = P.fwdb[c]; ba = P.bwdb[c]; ga = P.gateb[c];
      for (int k = 0; k < 160; ++k) {
        float xv = xa[k];
        fa += xv * P.fwdW[k * 1000 + c];
        ba += xv * P.bwdW[k * 1000 + c];
        ga += xv * P.gateW[k * 1000 + c];
      }
      bwds[c] = ba;
    }
    fr[it] = fa; gr[it] = ga;
  }
  __syncthreads();
  for (int it = 0; it < 4; ++it) {
    int c = t + it * 256;
    if (c >= 1000) continue;
    float fwdv = (c > m + 5) ? 0.f : fr[it];
    int ridx = m + 5 - c;
    float rev = (ridx >= 0 && ridx < 1000) ? bwds[ridx] : 0.f;
    float gate = 1.f / (1.f + __expf(-gr[it]));
    out[(size_t)g * 1000 + c] = fmaxf(gate * fwdv + (1.f - gate) * rev, 0.f);
  }
}

// ---------------------------------------------------------------------------
// Host orchestration
// ---------------------------------------------------------------------------
extern "C" void kernel_launch(void* const* d_in, const int* in_sizes, int n_in,
                              void* d_out, int out_size, void* d_ws, size_t ws_size,
                              hipStream_t stream) {
  (void)n_in; (void)out_size; (void)ws_size;
  const int N = in_sizes[0] / 50;   // 20000
  const int E = in_sizes[1] / 2;    // 320000
  const int G = in_sizes[3];        // 128

  const float* x   = (const float*)d_in[0];
  const int*   ei  = (const int*)d_in[1];
  const float* tm  = (const float*)d_in[3];
  const int*   bix = (const int*)d_in[4];
  // params (leaf order: sorted-key recursive flatten of the params dict)
  const int P = 5;
  auto F = [&](int i) { return (const float*)d_in[i]; };
  struct ConvP { const float *kW, *kb, *qW, *qb, *sW, *sb, *vW, *vb; } cv[5];
  for (int l = 0; l < 5; ++l) {
    int b = P + 2 + 8 * l;
    cv[l] = {F(b), F(b + 1), F(b + 2), F(b + 3), F(b + 4), F(b + 5), F(b + 6), F(b + 7)};
  }
  HeadArgs H;
  H.bwdW = F(P + 0);  H.bwdb = F(P + 1);
  H.fwdW = F(P + 42); H.fwdb = F(P + 43);
  H.gateW = F(P + 44); H.gateb = F(P + 45);
  H.h1W = F(P + 46); H.h1b = F(P + 47);
  H.h2W = F(P + 48); H.h2b = F(P + 49);
  H.h3W = F(P + 50); H.h3b = F(P + 51);
  H.h4W = F(P + 52); H.h4b = F(P + 53);
  struct SkipP { const float *b1, *b2, *g1, *g2, *v1, *v2; } sk[4];
  for (int s = 0; s < 4; ++s) {
    int b = P + 54 + 6 * s;
    sk[s] = {F(b), F(b + 1), F(b + 2), F(b + 3), F(b + 4), F(b + 5)};
  }
  H.s1v1 = sk[0].v1; H.s1b1 = sk[0].b1; H.s1v2 = sk[0].v2; H.s1b2 = sk[0].b2;
  H.s2v1 = sk[1].v1; H.s2b1 = sk[1].b1; H.s2v2 = sk[1].v2; H.s2b2 = sk[1].b2;
  H.s3v1 = sk[2].v1; H.s3b1 = sk[2].b1; H.s3v2 = sk[2].v2; H.s3b2 = sk[2].b2;
  H.s4v1 = sk[3].v1; H.s4b1 = sk[3].b1; H.s4v2 = sk[3].v2; H.s4b2 = sk[3].b2;

  // Workspace carve (256B aligned regions)
  char* ws = (char*)d_ws; size_t off = 0;
  auto carve = [&](size_t bytes) -> void* {
    void* p = ws + off; off = (off + bytes + 255) & ~(size_t)255; return p;
  };
  __bf16* Xb = (__bf16*)carve((size_t)N * 64 * 2);
  __bf16* hb = (__bf16*)carve((size_t)N * 160 * 2);
  __bf16* Wb[5];
  Wb[0] = (__bf16*)carve((size_t)64 * 640 * 2);
  for (int l = 1; l < 5; ++l) Wb[l] = (__bf16*)carve((size_t)160 * 640 * 2);
  float* bias640[5];
  for (int l = 0; l < 5; ++l) bias640[l] = (float*)carve(640 * 4);
  float* qkvs  = (float*)carve((size_t)N * 640 * 4);
  float* hf    = (float*)carve((size_t)N * 160 * 4);
  int* counts  = (int*)carve((size_t)N * 4);
  int* rowptr  = (int*)carve((size_t)(N + 1) * 4);
  int* cursor  = (int*)carve((size_t)N * 4);
  int* csr     = (int*)carve((size_t)E * 4);
  int* pool    = (int*)carve((size_t)G * 160 * 4);
  const int scCols[8] = {240, 160, 160, 240, 160, 320, 160, 160};
  const int scRows[8] = {160, 240, 240, 160, 320, 160, 160, 160};
  float* sc[8];
  for (int i = 0; i < 8; ++i) sc[i] = (float*)carve(scCols[i] * 4);
  H.sc1a = sc[0]; H.sc1b = sc[1]; H.sc2a = sc[2]; H.sc2b = sc[3];
  H.sc3a = sc[4]; H.sc3b = sc[5]; H.sc4a = sc[6]; H.sc4b = sc[7];

  // --- CSR build (edge_index constant; done per launch for determinism of buffers)
  zero_i32_kernel<<<cdiv(N, 256), 256, 0, stream>>>(counts, N);
  zero_i32_kernel<<<cdiv(G * 160, 256), 256, 0, stream>>>(pool, G * 160);
  count_deg_kernel<<<cdiv(E, 256), 256, 0, stream>>>(ei, E, counts);
  scan_block_kernel<<<1, 1024, 0, stream>>>(counts, rowptr, N);
  copy_i32_kernel<<<cdiv(N, 256), 256, 0, stream>>>(rowptr, cursor, N);
  csr_fill_kernel<<<cdiv(E, 256), 256, 0, stream>>>(ei, E, cursor, csr);

  // --- Pack inputs/weights to bf16 fragment layouts
  pack_x_kernel<<<cdiv(N * 64, 256), 256, 0, stream>>>(x, Xb, N * 64);
  for (int l = 0; l < 5; ++l) {
    int fin  = (l == 0) ? 50 : 160;
    int kpad = (l == 0) ? 64 : 160;
    pack_w_kernel<<<cdiv(kpad * 640, 256), 256, 0, stream>>>(
        cv[l].qW, cv[l].kW, cv[l].vW, cv[l].sW, Wb[l], fin, kpad * 640);
    pack_bias_kernel<<<cdiv(640, 256), 256, 0, stream>>>(
        cv[l].qb, cv[l].kb, cv[l].vb, cv[l].sb, bias640[l]);
  }
  const float* wnV[8] = {sk[0].v1, sk[0].v2, sk[1].v1, sk[1].v2,
                         sk[2].v1, sk[2].v2, sk[3].v1, sk[3].v2};
  const float* wnG[8] = {sk[0].g1, sk[0].g2, sk[1].g1, sk[1].g2,
                         sk[2].g1, sk[2].g2, sk[3].g1, sk[3].g2};
  for (int i = 0; i < 8; ++i)
    wn_scale_kernel<<<cdiv(scCols[i], 64), 64, 0, stream>>>(wnV[i], wnG[i], sc[i],
                                                            scRows[i], scCols[i]);

  // --- 5 TransformerConv layers: WMMA GEMM (q|k|v|skip) then edge attention
  for (int l = 0; l < 5; ++l) {
    int K = (l == 0) ? 64 : 160;
    const __bf16* A = (l == 0) ? Xb : hb;
    int waves = (N / 16) * 10;  // (M/16) x (640/64)
    gemm_wmma_kernel<<<cdiv(waves * 32, 256), 256, 0, stream>>>(A, Wb[l], bias640[l],
                                                                qkvs, N, K);
    attn_kernel<<<cdiv(N * 32, 256), 256, 0, stream>>>(qkvs, rowptr, csr, hf, hb, N);
  }

  // --- Global max pool and per-graph MLP head
  pool_kernel<<<cdiv(N * 160, 256), 256, 0, stream>>>(hf, bix, pool, N * 160);
  head_kernel<<<G, 256, 0, stream>>>(H, pool, tm, (float*)d_out);
}
